// ResidualAttentionBlock_13761075216787
// MI455X (gfx1250) — matrix operand order repaired and checked by reading, not compiled, not run on hardware
//
#include <hip/hip_runtime.h>
#include <hip/hip_bf16.h>

// ---------------------------------------------------------------------------
// ResidualAttentionBlock for MI455X (gfx1250): all GEMMs via
// v_wmma_f32_16x16x32_bf16 (bf16 inputs, fp32 accumulation).
// Branch-free steady-state k-loops: every launch dimension divides its tile,
// conv1 uses a zero-padded input so tap-shifted loads need no predicates.
// ---------------------------------------------------------------------------

typedef __bf16 bf16_t;
typedef __attribute__((ext_vector_type(16))) __bf16 v16bf;
typedef __attribute__((ext_vector_type(8)))  __bf16 v8bf;
typedef __attribute__((ext_vector_type(8)))  float  v8f;

#define Bq   4
#define Sq   1024
#define Tq   1024
#define Dq   1024
#define Hq   16
#define DHq  2048
#define TOK  4096          // B*S == B*T
#define SP   1032          // S + 8 (padded rows per batch for conv1)
#define KKc  9216          // 9 * 1024 conv1 unrolled K

// ------------------------- workspace layout (bytes) ------------------------
#define MB (1024LL*1024LL)
#define OFF_BFX     (0*MB)      // [4096,1024] bf16  (LN output / generic A)
#define OFF_BFQ     (8*MB)
#define OFF_BFK     (16*MB)
#define OFF_BFV     (24*MB)
#define OFF_BFVT    (32*MB)     // [B,H,64,T] bf16
#define OFF_PROBS   (40*MB)     // [B,H,S,T] bf16  (128 MB)
#define OFF_CTX     (168*MB)
#define OFF_HID     (176*MB)    // [4096,2048] bf16 (16 MB)
#define OFF_CONVIN  (192*MB)    // [B,1032,1024] bf16 padded (~8.06 MB)
#define OFF_BFXA    (202*MB)
#define OFF_X1      (210*MB)    // fp32 [4096,1024]
#define OFF_X2      (226*MB)
#define OFF_Y       (242*MB)
#define OFF_WQ1     (258*MB)    // 8 weight matrices, 2 MB each
#define OFF_WK1     (260*MB)
#define OFF_WV1     (262*MB)
#define OFF_WO1     (264*MB)
#define OFF_WQ2     (266*MB)
#define OFF_WK2     (268*MB)
#define OFF_WV2     (270*MB)
#define OFF_WO2     (272*MB)
#define OFF_WC2     (274*MB)    // [1024,2048] bf16 (4 MB)
#define OFF_WC1T    (278*MB)    // [2048, 9*1024] bf16 (~37 MB)

// --------------------------- fragment builders -----------------------------
// A fragment (16x32 bf16): lanes 0-15 M=lane K=k..k+15;
// lanes 16-31 K=k+16..k+31.  p points at row + half*16.
__device__ __forceinline__ v16bf frag_a_at(const bf16_t* __restrict__ p) {
  const v8bf* pv = reinterpret_cast<const v8bf*>(p);
  v8bf lo = pv[0];
  v8bf hi = pv[1];            // +8 elements
  v16bf f;
#pragma unroll
  for (int i = 0; i < 8; ++i) { f[i] = lo[i]; f[i + 8] = hi[i]; }
  return f;
}

// B fragment (32x16 bf16, source is B^T row-major): lanes 0-15 N=lane
// K=k..k+15; lanes 16-31 K=k+16..k+31.  p points at row + half*16.
__device__ __forceinline__ v16bf frag_b_at(const bf16_t* __restrict__ p) {
  const v8bf* pv = reinterpret_cast<const v8bf*>(p);
  v8bf lo = pv[0];
  v8bf hi = pv[1];            // +8 elements
  v16bf f;
#pragma unroll
  for (int i = 0; i < 8; ++i) { f[i] = lo[i]; f[i + 8] = hi[i]; }
  return f;
}

// ------------------------------- WMMA GEMM ---------------------------------
// C[M,N] = alpha * A[M,K] * B^T  (B stored as [N,K]) (+bias)(+residual)(relu)
// 256 threads = 8 waves; wave tile 32x64 (2x4 of 16x16).
// MW x NW wave grid (MW*NW==8): block tile = (MW*32) x (NW*64).
// All of M, N, K must divide the tile sizes (guaranteed by launcher).
template <int MW, int NW>
__global__ __launch_bounds__(256) void wmma_gemm_fast(
    const bf16_t* __restrict__ A, long long sA0, long long sA1, int lda,
    const bf16_t* __restrict__ Bm, long long sB0, long long sB1, int ldb,
    const float* __restrict__ bias,
    const float* __restrict__ Res, long long sR0, long long sR1, int ldr,
    float* __restrict__ outF, long long sF0, long long sF1, int ldof,
    bf16_t* __restrict__ outB, long long sOB0, long long sOB1, int ldob,
    int K, int zdiv, float alpha, int relu) {
  const int z  = blockIdx.z;
  const int z0 = z / zdiv, z1 = z % zdiv;
  A  += z0 * sA0 + z1 * sA1;
  Bm += z0 * sB0 + z1 * sB1;

  const int lane = threadIdx.x & 31;
  const int wid  = threadIdx.x >> 5;
  const int mBase = blockIdx.y * (MW * 32) + (wid % MW) * 32;
  const int nBase = blockIdx.x * (NW * 64) + (wid / MW) * 64;
  const int l15  = lane & 15;
  const int half = lane >> 4;

  const bf16_t* pa0 = A + (long long)(mBase + l15) * lda + half * 16;
  const bf16_t* pa1 = pa0 + (long long)16 * lda;
  const bf16_t* pb0 = Bm + (long long)(nBase + l15) * ldb + half * 16;
  const bf16_t* pb1 = pb0 + (long long)16 * ldb;
  const bf16_t* pb2 = pb0 + (long long)32 * ldb;
  const bf16_t* pb3 = pb0 + (long long)48 * ldb;

  v8f acc[2][4];
#pragma unroll
  for (int i = 0; i < 2; ++i)
#pragma unroll
    for (int j = 0; j < 4; ++j) acc[i][j] = (v8f){};

#pragma unroll 2
  for (int k0 = 0; k0 < K; k0 += 32) {
    v16bf a0 = frag_a_at(pa0 + k0);
    v16bf a1 = frag_a_at(pa1 + k0);
    v16bf b0 = frag_b_at(pb0 + k0);
    v16bf b1 = frag_b_at(pb1 + k0);
    v16bf b2 = frag_b_at(pb2 + k0);
    v16bf b3 = frag_b_at(pb3 + k0);
    acc[0][0] = __builtin_amdgcn_wmma_f32_16x16x32_bf16(false, a0, false, b0, (short)0, acc[0][0], false, false);
    acc[1][0] = __builtin_amdgcn_wmma_f32_16x16x32_bf16(false, a1, false, b0, (short)0, acc[1][0], false, false);
    acc[0][1] = __builtin_amdgcn_wmma_f32_16x16x32_bf16(false, a0, false, b1, (short)0, acc[0][1], false, false);
    acc[1][1] = __builtin_amdgcn_wmma_f32_16x16x32_bf16(false, a1, false, b1, (short)0, acc[1][1], false, false);
    acc[0][2] = __builtin_amdgcn_wmma_f32_16x16x32_bf16(false, a0, false, b2, (short)0, acc[0][2], false, false);
    acc[1][2] = __builtin_amdgcn_wmma_f32_16x16x32_bf16(false, a1, false, b2, (short)0, acc[1][2], false, false);
    acc[0][3] = __builtin_amdgcn_wmma_f32_16x16x32_bf16(false, a0, false, b3, (short)0, acc[0][3], false, false);
    acc[1][3] = __builtin_amdgcn_wmma_f32_16x16x32_bf16(false, a1, false, b3, (short)0, acc[1][3], false, false);
  }

  // ------------------------------ epilogue --------------------------------
  const float* ResZ = Res  ? Res  + z0 * sR0  + z1 * sR1  : nullptr;
  float*       outZ = outF ? outF + z0 * sF0  + z1 * sF1  : nullptr;
  bf16_t*      obZ  = outB ? outB + z0 * sOB0 + z1 * sOB1 : nullptr;
#pragma unroll
  for (int i = 0; i < 2; ++i) {
#pragma unroll
    for (int j = 0; j < 4; ++j) {
      int col = nBase + j * 16 + l15;
      float bv = bias ? bias[col] : 0.0f;
#pragma unroll
      for (int r = 0; r < 8; ++r) {
        int row = mBase + i * 16 + r + half * 8;
        float v = acc[i][j][r] * alpha + bv;
        if (ResZ) v += ResZ[(long long)row * ldr + col];
        if (relu) v = fmaxf(v, 0.0f);
        if (outZ) outZ[(long long)row * ldof + col] = v;
        if (obZ)  obZ[(long long)row * ldob + col] = (bf16_t)v;
      }
    }
  }
}

// Conv1: K = 9 taps x 1024, zero-padded input [B][1032][1024] so the shifted
// A loads are unconditional.  ReLU epilogue, bf16 out [4096,2048].
__global__ __launch_bounds__(256) void wmma_conv1_fast(
    const bf16_t* __restrict__ Xp,     // padded [B*1032, 1024] bf16
    const bf16_t* __restrict__ W,      // [2048, 9*1024] bf16 (co,tap,k)
    const float* __restrict__ bias,    // [2048]
    bf16_t* __restrict__ outB) {       // [4096,2048]
  const int lane = threadIdx.x & 31;
  const int wid  = threadIdx.x >> 5;
  const int mBase = blockIdx.y * 128 + (wid & 3) * 32;
  const int nBase = blockIdx.x * 128 + (wid >> 2) * 64;
  const int l15  = lane & 15;
  const int half = lane >> 4;

  // padded row for token m at tap t is m + 8*(m>>10) + t; all 32 rows of the
  // wave tile share one batch (1024 % 32 == 0).
  const int prow = mBase + 8 * (mBase >> 10);
  const bf16_t* pa0 = Xp + (long long)(prow + l15) * Dq + half * 16;
  const bf16_t* pa1 = pa0 + (long long)16 * Dq;
  const bf16_t* pb0 = W + (long long)(nBase + l15) * KKc + half * 16;
  const bf16_t* pb1 = pb0 + (long long)16 * KKc;
  const bf16_t* pb2 = pb0 + (long long)32 * KKc;
  const bf16_t* pb3 = pb0 + (long long)48 * KKc;

  v8f acc[2][4];
#pragma unroll
  for (int i = 0; i < 2; ++i)
#pragma unroll
    for (int j = 0; j < 4; ++j) acc[i][j] = (v8f){};

  for (int tap = 0; tap < 9; ++tap) {
    const bf16_t* at0 = pa0 + (long long)tap * Dq;
    const bf16_t* at1 = pa1 + (long long)tap * Dq;
    const long long wk = (long long)tap * Dq;
#pragma unroll 2
    for (int k0 = 0; k0 < Dq; k0 += 32) {
      v16bf a0 = frag_a_at(at0 + k0);
      v16bf a1 = frag_a_at(at1 + k0);
      v16bf b0 = frag_b_at(pb0 + wk + k0);
      v16bf b1 = frag_b_at(pb1 + wk + k0);
      v16bf b2 = frag_b_at(pb2 + wk + k0);
      v16bf b3 = frag_b_at(pb3 + wk + k0);
      acc[0][0] = __builtin_amdgcn_wmma_f32_16x16x32_bf16(false, a0, false, b0, (short)0, acc[0][0], false, false);
      acc[1][0] = __builtin_amdgcn_wmma_f32_16x16x32_bf16(false, a1, false, b0, (short)0, acc[1][0], false, false);
      acc[0][1] = __builtin_amdgcn_wmma_f32_16x16x32_bf16(false, a0, false, b1, (short)0, acc[0][1], false, false);
      acc[1][1] = __builtin_amdgcn_wmma_f32_16x16x32_bf16(false, a1, false, b1, (short)0, acc[1][1], false, false);
      acc[0][2] = __builtin_amdgcn_wmma_f32_16x16x32_bf16(false, a0, false, b2, (short)0, acc[0][2], false, false);
      acc[1][2] = __builtin_amdgcn_wmma_f32_16x16x32_bf16(false, a1, false, b2, (short)0, acc[1][2], false, false);
      acc[0][3] = __builtin_amdgcn_wmma_f32_16x16x32_bf16(false, a0, false, b3, (short)0, acc[0][3], false, false);
      acc[1][3] = __builtin_amdgcn_wmma_f32_16x16x32_bf16(false, a1, false, b3, (short)0, acc[1][3], false, false);
    }
  }

#pragma unroll
  for (int i = 0; i < 2; ++i)
#pragma unroll
    for (int j = 0; j < 4; ++j) {
      int col = nBase + j * 16 + l15;
      float bv = bias[col];
#pragma unroll
      for (int r = 0; r < 8; ++r) {
        int row = mBase + i * 16 + r + half * 8;
        float v = fmaxf(acc[i][j][r] + bv, 0.0f);
        outB[(long long)row * DHq + col] = (bf16_t)v;
      }
    }
}

// ----------------------------- LayerNorm -----------------------------------
__global__ __launch_bounds__(256) void layernorm_kernel(
    const float* __restrict__ x, const float* __restrict__ g,
    const float* __restrict__ b, bf16_t* __restrict__ outB,
    float* __restrict__ outF, int rows) {
  int row  = blockIdx.x * 8 + (threadIdx.x >> 5);
  int lane = threadIdx.x & 31;
  if (row >= rows) return;
  const float4* xp = reinterpret_cast<const float4*>(x + (long long)row * Dq);
  float4 v[8];
  float s = 0.0f, s2 = 0.0f;
#pragma unroll
  for (int i = 0; i < 8; ++i) {
    v[i] = xp[lane + 32 * i];
    s  += v[i].x + v[i].y + v[i].z + v[i].w;
    s2 += v[i].x * v[i].x + v[i].y * v[i].y + v[i].z * v[i].z + v[i].w * v[i].w;
  }
#pragma unroll
  for (int m = 16; m >= 1; m >>= 1) {
    s  += __shfl_xor(s,  m, 32);
    s2 += __shfl_xor(s2, m, 32);
  }
  float mu  = s * (1.0f / Dq);
  float var = s2 * (1.0f / Dq) - mu * mu;
  float inv = rsqrtf(var + 1e-5f);
  const float4* gp = reinterpret_cast<const float4*>(g);
  const float4* bp = reinterpret_cast<const float4*>(b);
#pragma unroll
  for (int i = 0; i < 8; ++i) {
    int c4 = lane + 32 * i;
    float4 gg = gp[c4], bb = bp[c4];
    float4 y;
    y.x = (v[i].x - mu) * inv * gg.x + bb.x;
    y.y = (v[i].y - mu) * inv * gg.y + bb.y;
    y.z = (v[i].z - mu) * inv * gg.z + bb.z;
    y.w = (v[i].w - mu) * inv * gg.w + bb.w;
    if (outF) reinterpret_cast<float4*>(outF + (long long)row * Dq)[c4] = y;
    if (outB) {
      bf16_t* ob = outB + (long long)row * Dq + c4 * 4;
      ob[0] = (bf16_t)y.x; ob[1] = (bf16_t)y.y;
      ob[2] = (bf16_t)y.z; ob[3] = (bf16_t)y.w;
    }
  }
}

// ----------------------------- Softmax -------------------------------------
__global__ __launch_bounds__(256) void softmax_kernel(
    float* __restrict__ scores, bf16_t* __restrict__ probs, long long rows) {
  long long row = (long long)blockIdx.x * 8 + (threadIdx.x >> 5);
  int lane = threadIdx.x & 31;
  if (row >= rows) return;
  float4* p = reinterpret_cast<float4*>(scores + row * (long long)Tq);
  float4 v[8];
  float mx = -3.0e38f;
#pragma unroll
  for (int i = 0; i < 8; ++i) {
    v[i] = p[lane + 32 * i];
    mx = fmaxf(mx, fmaxf(fmaxf(v[i].x, v[i].y), fmaxf(v[i].z, v[i].w)));
  }
#pragma unroll
  for (int m = 16; m >= 1; m >>= 1) mx = fmaxf(mx, __shfl_xor(mx, m, 32));
  float sum = 0.0f;
#pragma unroll
  for (int i = 0; i < 8; ++i) {
    v[i].x = __expf(v[i].x - mx); v[i].y = __expf(v[i].y - mx);
    v[i].z = __expf(v[i].z - mx); v[i].w = __expf(v[i].w - mx);
    sum += v[i].x + v[i].y + v[i].z + v[i].w;
  }
#pragma unroll
  for (int m = 16; m >= 1; m >>= 1) sum += __shfl_xor(sum, m, 32);
  float inv = 1.0f / sum;
#pragma unroll
  for (int i = 0; i < 8; ++i) {
    v[i].x *= inv; v[i].y *= inv; v[i].z *= inv; v[i].w *= inv;
    p[lane + 32 * i] = v[i];
    bf16_t* ob = probs + row * (long long)Tq + (lane + 32 * i) * 4;
    ob[0] = (bf16_t)v[i].x; ob[1] = (bf16_t)v[i].y;
    ob[2] = (bf16_t)v[i].z; ob[3] = (bf16_t)v[i].w;
  }
}

// ------------------------- small utility kernels ---------------------------
__global__ void castf2b_kernel(const float* __restrict__ src,
                               bf16_t* __restrict__ dst, long long n4) {
  long long i = (long long)blockIdx.x * blockDim.x + threadIdx.x;
  if (i >= n4) return;
  float4 v = reinterpret_cast<const float4*>(src)[i];
  bf16_t* d = dst + i * 4;
  d[0] = (bf16_t)v.x; d[1] = (bf16_t)v.y; d[2] = (bf16_t)v.z; d[3] = (bf16_t)v.w;
}

// f_x2 [B*S, D] fp32 -> padded conv input [B][S+8][D] bf16 (4 zero rows each
// side of every batch's sequence).
__global__ void pad_cast_kernel(const float* __restrict__ src,
                                bf16_t* __restrict__ dst) {
  long long i = (long long)blockIdx.x * blockDim.x + threadIdx.x;  // B*SP*D
  if (i >= (long long)Bq * SP * Dq) return;
  int c = (int)(i % Dq);
  int r = (int)((i / Dq) % SP);
  long long b = i / ((long long)SP * Dq);
  float v = 0.0f;
  if (r >= 4 && r < SP - 4)
    v = src[(b * Sq + (r - 4)) * Dq + c];
  dst[i] = (bf16_t)v;
}

// w_c1 [DH, D, 9] fp32  ->  [DH, 9, D] bf16  (tap-major K for conv GEMM)
__global__ void convw_kernel(const float* __restrict__ src, bf16_t* __restrict__ dst) {
  long long i = (long long)blockIdx.x * blockDim.x + threadIdx.x;  // over DH*9*D
  if (i >= (long long)DHq * 9 * Dq) return;
  int k   = (int)(i % Dq);
  int t   = (int)((i / Dq) % 9);
  long long co = i / ((long long)9 * Dq);
  dst[i] = (bf16_t)src[(co * Dq + k) * 9 + t];
}

// bf_v [B*T, D] -> vT [B,H,64,T]  (so PV GEMM's B operand is lane-contiguous)
__global__ void transpose_v_kernel(const bf16_t* __restrict__ v,
                                   bf16_t* __restrict__ vT) {
  long long i = (long long)blockIdx.x * blockDim.x + threadIdx.x;  // B*H*64*T
  if (i >= (long long)Bq * Hq * 64 * Tq) return;
  int t = (int)(i & (Tq - 1));
  long long rest = i >> 10;
  int d = (int)(rest & 63);
  long long bh = rest >> 6;
  int h = (int)(bh & (Hq - 1));
  long long b = bh >> 4;
  vT[i] = v[((b * Tq + t) * Dq) + h * 64 + d];
}

// ------------------------------ launcher -----------------------------------
static inline void gemm128(hipStream_t st, const bf16_t* A, int lda,
                           const bf16_t* B, int ldb, const float* bias,
                           const float* Res, int ldr, float* outF, int ldof,
                           bf16_t* outB, int ldob, int M, int N, int K,
                           float alpha, int relu) {
  dim3 g(N / 128, M / 128, 1);
  wmma_gemm_fast<4, 2><<<g, 256, 0, st>>>(A, 0, 0, lda, B, 0, 0, ldb, bias,
                                          Res, 0, 0, ldr, outF, 0, 0, ldof,
                                          outB, 0, 0, ldob, K, 1, alpha, relu);
}

extern "C" void kernel_launch(void* const* d_in, const int* in_sizes, int n_in,
                              void* d_out, int out_size, void* d_ws, size_t ws_size,
                              hipStream_t stream) {
  const float* x    = (const float*)d_in[0];
  const float* xa   = (const float*)d_in[1];
  const float* Wq1  = (const float*)d_in[2];
  const float* bq1  = (const float*)d_in[3];
  const float* Wk1  = (const float*)d_in[4];
  const float* Wv1  = (const float*)d_in[5];
  const float* bv1  = (const float*)d_in[6];
  const float* Wo1  = (const float*)d_in[7];
  const float* bo1  = (const float*)d_in[8];
  const float* gln1 = (const float*)d_in[9];
  const float* bln1 = (const float*)d_in[10];
  const float* Wq2  = (const float*)d_in[11];
  const float* bq2  = (const float*)d_in[12];
  const float* Wk2  = (const float*)d_in[13];
  const float* Wv2  = (const float*)d_in[14];
  const float* bv2  = (const float*)d_in[15];
  const float* Wo2  = (const float*)d_in[16];
  const float* bo2  = (const float*)d_in[17];
  const float* gln2 = (const float*)d_in[18];
  const float* bln2 = (const float*)d_in[19];
  const float* w_c1 = (const float*)d_in[20];
  const float* b_c1 = (const float*)d_in[21];
  const float* w_c2 = (const float*)d_in[22];
  const float* b_c2 = (const float*)d_in[23];
  const float* gln3 = (const float*)d_in[24];
  const float* bln3 = (const float*)d_in[25];

  char* ws = (char*)d_ws;
  bf16_t* bf_x    = (bf16_t*)(ws + OFF_BFX);
  bf16_t* bf_q    = (bf16_t*)(ws + OFF_BFQ);
  bf16_t* bf_k    = (bf16_t*)(ws + OFF_BFK);
  bf16_t* bf_v    = (bf16_t*)(ws + OFF_BFV);
  bf16_t* bf_vT   = (bf16_t*)(ws + OFF_BFVT);
  bf16_t* bf_pr   = (bf16_t*)(ws + OFF_PROBS);
  bf16_t* bf_ctx  = (bf16_t*)(ws + OFF_CTX);
  bf16_t* bf_hid  = (bf16_t*)(ws + OFF_HID);
  bf16_t* bf_ci   = (bf16_t*)(ws + OFF_CONVIN);
  bf16_t* bf_xa   = (bf16_t*)(ws + OFF_BFXA);
  float*  f_x1    = (float*)(ws + OFF_X1);
  float*  f_x2    = (float*)(ws + OFF_X2);
  float*  f_y     = (float*)(ws + OFF_Y);
  bf16_t* bWq1 = (bf16_t*)(ws + OFF_WQ1);
  bf16_t* bWk1 = (bf16_t*)(ws + OFF_WK1);
  bf16_t* bWv1 = (bf16_t*)(ws + OFF_WV1);
  bf16_t* bWo1 = (bf16_t*)(ws + OFF_WO1);
  bf16_t* bWq2 = (bf16_t*)(ws + OFF_WQ2);
  bf16_t* bWk2 = (bf16_t*)(ws + OFF_WK2);
  bf16_t* bWv2 = (bf16_t*)(ws + OFF_WV2);
  bf16_t* bWo2 = (bf16_t*)(ws + OFF_WO2);
  bf16_t* bWc2 = (bf16_t*)(ws + OFF_WC2);
  bf16_t* bWc1 = (bf16_t*)(ws + OFF_WC1T);

  float* out_x  = (float*)d_out;
  float* attn1  = out_x + (long long)Bq * Sq * Dq;                 // 4M floats
  float* attn2  = attn1 + (long long)Bq * Hq * Sq * Tq;            // +64M floats

  const long long DD4 = (long long)Dq * Dq / 4;
  castf2b_kernel<<<(unsigned)((DD4 + 255) / 256), 256, 0, stream>>>(Wq1, bWq1, DD4);
  castf2b_kernel<<<(unsigned)((DD4 + 255) / 256), 256, 0, stream>>>(Wk1, bWk1, DD4);
  castf2b_kernel<<<(unsigned)((DD4 + 255) / 256), 256, 0, stream>>>(Wv1, bWv1, DD4);
  castf2b_kernel<<<(unsigned)((DD4 + 255) / 256), 256, 0, stream>>>(Wo1, bWo1, DD4);
  castf2b_kernel<<<(unsigned)((DD4 + 255) / 256), 256, 0, stream>>>(Wq2, bWq2, DD4);
  castf2b_kernel<<<(unsigned)((DD4 + 255) / 256), 256, 0, stream>>>(Wk2, bWk2, DD4);
  castf2b_kernel<<<(unsigned)((DD4 + 255) / 256), 256, 0, stream>>>(Wv2, bWv2, DD4);
  castf2b_kernel<<<(unsigned)((DD4 + 255) / 256), 256, 0, stream>>>(Wo2, bWo2, DD4);
  const long long WC24 = (long long)Dq * DHq / 4;
  castf2b_kernel<<<(unsigned)((WC24 + 255) / 256), 256, 0, stream>>>(w_c2, bWc2, WC24);
  const long long XA4 = (long long)TOK * Dq / 4;
  castf2b_kernel<<<(unsigned)((XA4 + 255) / 256), 256, 0, stream>>>(xa, bf_xa, XA4);
  const long long CW = (long long)DHq * 9 * Dq;
  convw_kernel<<<(unsigned)((CW + 255) / 256), 256, 0, stream>>>(w_c1, bWc1);

  const float attn_scale = 0.125f;   // 1/sqrt(dh) = 1/8
  const long long SD = (long long)Sq * Dq;   // per-batch token stride
  const long long ST = (long long)Sq * Tq;   // per-head score stride
  const unsigned TRN = (unsigned)(((long long)Bq * Hq * 64 * Tq + 255) / 256);

  // ============================ self-attention ============================
  layernorm_kernel<<<TOK / 8, 256, 0, stream>>>(x, gln1, bln1, bf_x, nullptr, TOK);
  gemm128(stream, bf_x, Dq, bWq1, Dq, bq1, nullptr, 0, nullptr, 0,
          bf_q, Dq, TOK, Dq, Dq, 1.0f, 0);
  gemm128(stream, bf_x, Dq, bWk1, Dq, nullptr, nullptr, 0, nullptr, 0,
          bf_k, Dq, TOK, Dq, Dq, 1.0f, 0);
  gemm128(stream, bf_x, Dq, bWv1, Dq, bv1, nullptr, 0, nullptr, 0,
          bf_v, Dq, TOK, Dq, Dq, 1.0f, 0);
  transpose_v_kernel<<<TRN, 256, 0, stream>>>(bf_v, bf_vT);
  {
    dim3 g(Tq / 128, Sq / 128, Bq * Hq);
    wmma_gemm_fast<4, 2><<<g, 256, 0, stream>>>(
        bf_q, SD, 64, Dq, bf_k, SD, 64, Dq, nullptr,
        nullptr, 0, 0, 0,
        attn1, (long long)Hq * ST, ST, Tq,
        nullptr, 0, 0, 0, 64, Hq, attn_scale, 0);
  }
  softmax_kernel<<<(unsigned)((long long)Bq * Hq * Sq / 8), 256, 0, stream>>>(
      attn1, bf_pr, (long long)Bq * Hq * Sq);
  {
    dim3 g(1, Sq / 256, Bq * Hq);   // 256x64 tile: 8x1 wave grid, N=64 exact
    wmma_gemm_fast<8, 1><<<g, 256, 0, stream>>>(
        bf_pr, (long long)Hq * ST, ST, Tq,
        bf_vT, (long long)Hq * 64 * Tq, (long long)64 * Tq, Tq, nullptr,
        nullptr, 0, 0, 0, nullptr, 0, 0, 0,
        bf_ctx, SD, 64, Dq, Tq, Hq, 1.0f, 0);
  }
  gemm128(stream, bf_ctx, Dq, bWo1, Dq, bo1, x, Dq, f_x1, Dq,
          nullptr, 0, TOK, Dq, Dq, 1.0f, 0);

  // ============================ cross-attention ===========================
  layernorm_kernel<<<TOK / 8, 256, 0, stream>>>(f_x1, gln2, bln2, bf_x, nullptr, TOK);
  gemm128(stream, bf_x,  Dq, bWq2, Dq, bq2, nullptr, 0, nullptr, 0,
          bf_q, Dq, TOK, Dq, Dq, 1.0f, 0);
  gemm128(stream, bf_xa, Dq, bWk2, Dq, nullptr, nullptr, 0, nullptr, 0,
          bf_k, Dq, TOK, Dq, Dq, 1.0f, 0);
  gemm128(stream, bf_xa, Dq, bWv2, Dq, bv2, nullptr, 0, nullptr, 0,
          bf_v, Dq, TOK, Dq, Dq, 1.0f, 0);
  transpose_v_kernel<<<TRN, 256, 0, stream>>>(bf_v, bf_vT);
  {
    dim3 g(Tq / 128, Sq / 128, Bq * Hq);
    wmma_gemm_fast<4, 2><<<g, 256, 0, stream>>>(
        bf_q, SD, 64, Dq, bf_k, SD, 64, Dq, nullptr,
        nullptr, 0, 0, 0,
        attn2, (long long)Hq * ST, ST, Tq,
        nullptr, 0, 0, 0, 64, Hq, attn_scale, 0);
  }
  softmax_kernel<<<(unsigned)((long long)Bq * Hq * Sq / 8), 256, 0, stream>>>(
      attn2, bf_pr, (long long)Bq * Hq * Sq);
  {
    dim3 g(1, Sq / 256, Bq * Hq);
    wmma_gemm_fast<8, 1><<<g, 256, 0, stream>>>(
        bf_pr, (long long)Hq * ST, ST, Tq,
        bf_vT, (long long)Hq * 64 * Tq, (long long)64 * Tq, Tq, nullptr,
        nullptr, 0, 0, 0, nullptr, 0, 0, 0,
        bf_ctx, SD, 64, Dq, Tq, Hq, 1.0f, 0);
  }
  gemm128(stream, bf_ctx, Dq, bWo2, Dq, bo2, f_x1, Dq, f_x2, Dq,
          nullptr, 0, TOK, Dq, Dq, 1.0f, 0);

  // ============================== conv FFN ================================
  const long long PADN = (long long)Bq * SP * Dq;
  pad_cast_kernel<<<(unsigned)((PADN + 255) / 256), 256, 0, stream>>>(f_x2, bf_ci);
  {
    dim3 g(DHq / 128, TOK / 128, 1);
    wmma_conv1_fast<<<g, 256, 0, stream>>>(bf_ci, bWc1, b_c1, bf_hid);
  }
  gemm128(stream, bf_hid, DHq, bWc2, DHq, b_c2, f_x2, Dq, f_y, Dq,
          nullptr, 0, TOK, Dq, DHq, 1.0f, 0);
  layernorm_kernel<<<TOK / 8, 256, 0, stream>>>(f_y, gln3, bln3, nullptr, out_x, TOK);
}
